// JointBPBook_5841155522735
// MI455X (gfx1250) — matrix operand order repaired and checked by reading, not compiled, not run on hardware
//
#include <hip/hip_runtime.h>
#include <hip/hip_bf16.h>
#include <math.h>

#define B_    16384
#define N_    64
#define D_    128
#define S_    16384
#define TOPK_ 5
#define EPS_  1e-12f

#define TS_   32                    // slots (columns) per pipeline stage
#define NT_   (S_ / TS_)            // 512 stages
#define TBYTES_ (TS_ * D_ * 2)      // 8 KB of bf16 B-tile per stage
#define ROWS_PER_WG_ 64             // 4 waves x 16 rows

typedef __attribute__((ext_vector_type(16))) __bf16        v16bf;
typedef __attribute__((ext_vector_type(8)))  float         v8f;
typedef __attribute__((ext_vector_type(4)))  unsigned int  v4u;
typedef __attribute__((ext_vector_type(4)))  int           v4i;

union Frag32 { v16bf v; v4u u[2]; };

#if __has_builtin(__builtin_amdgcn_global_load_async_to_lds_b128)
#define ASYNC_LDS_ 1
typedef __attribute__((address_space(1))) v4i* g_v4ip;   // global int32x4*
typedef __attribute__((address_space(3))) v4i* l_v4ip;   // LDS    int32x4*
#endif

__device__ __forceinline__ unsigned short f32_to_bf16_rne(float f) {
  unsigned int u = __float_as_uint(f);
  unsigned int r = 0x7FFFu + ((u >> 16) & 1u);
  return (unsigned short)((u + r) >> 16);
}

// ---------------------------------------------------------------------------
// Kernel 1: query = l2norm(mean_n(x) @ Wq^T + bq) -> bf16
// ---------------------------------------------------------------------------
__global__ __launch_bounds__(128) void qprep_kernel(
    const float* __restrict__ x, const float* __restrict__ Wq,
    const float* __restrict__ bq, unsigned short* __restrict__ qbf) {
  const int b = blockIdx.x;
  const int t = threadIdx.x;
  __shared__ float qmean[D_];
  __shared__ float red[128];

  const float* xb = x + (size_t)b * (N_ * D_);
  float s = 0.f;
#pragma unroll 8
  for (int n = 0; n < N_; ++n) s += xb[n * D_ + t];
  qmean[t] = s * (1.0f / (float)N_);
  __syncthreads();

  const float* wrow = Wq + (size_t)t * D_;
  float acc = bq[t];
#pragma unroll 8
  for (int d = 0; d < D_; ++d) acc += qmean[d] * wrow[d];

  red[t] = acc * acc;
  __syncthreads();
  for (int off = 64; off > 0; off >>= 1) {
    if (t < off) red[t] += red[t + off];
    __syncthreads();
  }
  const float rinv = 1.0f / fmaxf(sqrtf(red[0]), EPS_);
  qbf[(size_t)b * D_ + t] = f32_to_bf16_rne(acc * rinv);
}

// ---------------------------------------------------------------------------
// Kernel 2: m_norm = l2norm(memory) -> bf16
// ---------------------------------------------------------------------------
__global__ __launch_bounds__(128) void mprep_kernel(
    const float* __restrict__ mem, unsigned short* __restrict__ mbf) {
  const int s = blockIdx.x;
  const int t = threadIdx.x;
  __shared__ float red[128];
  const float v = mem[(size_t)s * D_ + t];
  red[t] = v * v;
  __syncthreads();
  for (int off = 64; off > 0; off >>= 1) {
    if (t < off) red[t] += red[t + off];
    __syncthreads();
  }
  const float rinv = 1.0f / fmaxf(sqrtf(red[0]), EPS_);
  mbf[(size_t)s * D_ + t] = f32_to_bf16_rne(v * rinv);
}

// ---------------------------------------------------------------------------
// Kernel 3: 4 waves / 64 rows per WG; B tiles shared via double-buffered LDS
// (async global->LDS), two interleaved WMMA accumulator chains per stage,
// per-lane top-5, LDS merge, softmax, prototype gather.
// ---------------------------------------------------------------------------
__global__ __launch_bounds__(128) void simtopk_kernel(
    const unsigned short* __restrict__ qbf,
    const unsigned short* __restrict__ mbf,
    const float* __restrict__ mem,
    float* __restrict__ proto) {
  const int  tid      = threadIdx.x;
  const int  lane     = tid & 31;
  const int  w        = tid >> 5;                 // wave id 0..3
  const int  row_base = blockIdx.x * ROWS_PER_WG_ + w * 16;
  const int  m16      = lane & 15;
  const bool hi       = lane >= 16;

  __shared__ union alignas(32) {
    unsigned char bbuf[2][TBYTES_];               // main loop: B tiles
    struct {                                      // epilogue: top-k merge
      float s_val[4][16][16][TOPK_];
      int   s_idx[4][16][16][TOPK_];
      float s_w[4][16][TOPK_];
      int   s_i[4][16][TOPK_];
    } mg;
  } sm;

  // ---- A fragments: 16x128 query tile, contiguous k = 16h + s per lane half (B matches) -------------
  Frag32 a[4];
  {
    const unsigned short* qp = qbf + (size_t)(row_base + m16) * D_;
    const int k0 = hi ? 16 : 0;
#pragma unroll
    for (int t = 0; t < 4; ++t) {
      const unsigned short* p = qp + t * 32 + k0;
      a[t].u[0] = *(const v4u*)(p);
      a[t].u[1] = *(const v4u*)(p + 8);
    }
  }

  float tv[8][TOPK_];
  int   ti[8][TOPK_];
#pragma unroll
  for (int r = 0; r < 8; ++r)
#pragma unroll
    for (int j = 0; j < TOPK_; ++j) { tv[r][j] = -INFINITY; ti[r][j] = 0; }

  const unsigned char* gmb = (const unsigned char*)mbf;

  // ---- prologue: fill buffer 0 with stage 0 ------------------------------
#if ASYNC_LDS_
#pragma unroll
  for (int i = 0; i < 4; ++i) {
    const int off = (tid + i * 128) * 16;
    __builtin_amdgcn_global_load_async_to_lds_b128(
        (g_v4ip)(gmb + off), (l_v4ip)(&sm.bbuf[0][off]), 0, 0);
  }
#if __has_builtin(__builtin_amdgcn_s_wait_asynccnt)
  __builtin_amdgcn_s_wait_asynccnt(0);
#else
  asm volatile("s_wait_asynccnt 0x0" ::: "memory");
#endif
#else
  {
    v4u st[4];
#pragma unroll
    for (int i = 0; i < 4; ++i) st[i] = *(const v4u*)(gmb + (tid + i * 128) * 16);
#pragma unroll
    for (int i = 0; i < 4; ++i) *(v4u*)(&sm.bbuf[0][(tid + i * 128) * 16]) = st[i];
  }
#endif
  __syncthreads();

  // ---- pipelined main loop ----------------------------------------------
  for (int tile = 0; tile < NT_; ++tile) {
    const int cur = tile & 1;
    const bool have_next = (tile + 1) < NT_;
    const unsigned char* gnext = gmb + (size_t)(tile + 1) * TBYTES_;

#if ASYNC_LDS_
    if (have_next) {
#pragma unroll
      for (int i = 0; i < 4; ++i) {
        const int off = (tid + i * 128) * 16;
        __builtin_amdgcn_global_load_async_to_lds_b128(
            (g_v4ip)(gnext + off), (l_v4ip)(&sm.bbuf[cur ^ 1][off]), 0, 0);
      }
    }
#else
    v4u st[4];
    if (have_next) {
#pragma unroll
      for (int i = 0; i < 4; ++i) st[i] = *(const v4u*)(gnext + (tid + i * 128) * 16);
    }
#endif

    // ---- compute on bbuf[cur]: two independent WMMA chains ---------------
    // column tile 0: slots [tile*32, +16); column tile 1: slots [+16, +32)
    const unsigned char* bb0 = &sm.bbuf[cur][(m16)      * (D_ * 2) + (hi ? 32 : 0)];
    const unsigned char* bb1 = &sm.bbuf[cur][(16 + m16) * (D_ * 2) + (hi ? 32 : 0)];
    v8f cc0 = {}, cc1 = {};
    v16bf b0 = *(const v16bf*)(bb0);
    v16bf b1 = *(const v16bf*)(bb1);
#pragma unroll
    for (int t = 0; t < 4; ++t) {
      const v16bf c0 = b0, c1 = b1;
      if (t < 3) {                                 // hoist next K-step's loads
        b0 = *(const v16bf*)(bb0 + (t + 1) * 64);
        b1 = *(const v16bf*)(bb1 + (t + 1) * 64);
      }
      cc0 = __builtin_amdgcn_wmma_f32_16x16x32_bf16(
                false, a[t].v, false, c0, (short)0, cc0, false, false);
      cc1 = __builtin_amdgcn_wmma_f32_16x16x32_bf16(
                false, a[t].v, false, c1, (short)0, cc1, false, false);
    }

    // ---- top-5 updates for both column tiles -----------------------------
#pragma unroll
    for (int c = 0; c < 2; ++c) {
      const int slot = tile * TS_ + c * 16 + m16;
#pragma unroll
      for (int r = 0; r < 8; ++r) {
        float val = (c == 0) ? cc0[r] : cc1[r];
        if (val > tv[r][TOPK_ - 1]) {
          float v = val; int id = slot;
#pragma unroll
          for (int j = 0; j < TOPK_; ++j) {
            if (v > tv[r][j]) {
              float tf = tv[r][j]; tv[r][j] = v;  v  = tf;
              int   tg = ti[r][j]; ti[r][j] = id; id = tg;
            }
          }
        }
      }
    }

#if ASYNC_LDS_
    if (have_next) {
#if __has_builtin(__builtin_amdgcn_s_wait_asynccnt)
      __builtin_amdgcn_s_wait_asynccnt(0);
#else
      asm volatile("s_wait_asynccnt 0x0" ::: "memory");
#endif
    }
#else
    if (have_next) {
#pragma unroll
      for (int i = 0; i < 4; ++i)
        *(v4u*)(&sm.bbuf[cur ^ 1][(tid + i * 128) * 16]) = st[i];
    }
#endif
    __syncthreads();
  }

  // ---- merge 16 lane-classes per row, softmax ---------------------------
  {
    const int rowoff = hi ? 8 : 0;
#pragma unroll
    for (int r = 0; r < 8; ++r)
#pragma unroll
      for (int j = 0; j < TOPK_; ++j) {
        sm.mg.s_val[w][rowoff + r][m16][j] = tv[r][j];
        sm.mg.s_idx[w][rowoff + r][m16][j] = ti[r][j];
      }
  }
  __syncthreads();

  if (lane < 16) {
    float bv[TOPK_]; int bi[TOPK_];
#pragma unroll
    for (int j = 0; j < TOPK_; ++j) { bv[j] = -INFINITY; bi[j] = 0; }
    for (int c = 0; c < 16; ++c) {
#pragma unroll
      for (int j = 0; j < TOPK_; ++j) {
        float v = sm.mg.s_val[w][lane][c][j]; int id = sm.mg.s_idx[w][lane][c][j];
        if (v > bv[TOPK_ - 1]) {
#pragma unroll
          for (int jj = 0; jj < TOPK_; ++jj) {
            if (v > bv[jj]) {
              float tf = bv[jj]; bv[jj] = v;  v  = tf;
              int   tg = bi[jj]; bi[jj] = id; id = tg;
            }
          }
        }
      }
    }
    const float mx = bv[0];
    float e[TOPK_], sum = 0.f;
#pragma unroll
    for (int j = 0; j < TOPK_; ++j) { e[j] = __expf(bv[j] - mx); sum += e[j]; }
    const float rs = 1.0f / sum;
#pragma unroll
    for (int j = 0; j < TOPK_; ++j) {
      sm.mg.s_w[w][lane][j] = e[j] * rs;
      sm.mg.s_i[w][lane][j] = bi[j];
    }
  }
  __syncthreads();

  // ---- prototype gather: each wave writes its 16 rows --------------------
  const int d0 = lane * 4;
  for (int m = 0; m < 16; ++m) {
    float4 acc = make_float4(0.f, 0.f, 0.f, 0.f);
#pragma unroll
    for (int k = 0; k < TOPK_; ++k) {
      const float  wgt = sm.mg.s_w[w][m][k];
      const float4 mv  = *(const float4*)(mem + (size_t)sm.mg.s_i[w][m][k] * D_ + d0);
      acc.x += wgt * mv.x; acc.y += wgt * mv.y;
      acc.z += wgt * mv.z; acc.w += wgt * mv.w;
    }
    *(float4*)(proto + (size_t)(row_base + m) * D_ + d0) = acc;
  }
}

// ---------------------------------------------------------------------------
// Kernel 4: out = x + scale * proto[b, d]
// ---------------------------------------------------------------------------
__global__ __launch_bounds__(256) void addout_kernel(
    const float* __restrict__ x, const float* __restrict__ proto,
    const float* __restrict__ scale_p, float* __restrict__ out) {
  const size_t tid   = (size_t)blockIdx.x * blockDim.x + threadIdx.x;
  const size_t total = (size_t)B_ * N_ * (D_ / 4);
  if (tid >= total) return;
  const float  s  = scale_p[0];
  const int    d4 = (int)(tid & 31);
  const size_t b  = tid >> 11;
  const float4 xv = ((const float4*)x)[tid];
  const float4 pv = ((const float4*)proto)[b * 32 + d4];
  float4 o;
  o.x = xv.x + s * pv.x;  o.y = xv.y + s * pv.y;
  o.z = xv.z + s * pv.z;  o.w = xv.w + s * pv.w;
  ((float4*)out)[tid] = o;
}

// ---------------------------------------------------------------------------
extern "C" void kernel_launch(void* const* d_in, const int* in_sizes, int n_in,
                              void* d_out, int out_size, void* d_ws, size_t ws_size,
                              hipStream_t stream) {
  const float* x     = (const float*)d_in[0];
  const float* mem   = (const float*)d_in[1];
  const float* scale = (const float*)d_in[2];
  const float* Wq    = (const float*)d_in[3];
  const float* bq    = (const float*)d_in[4];
  float*       out   = (float*)d_out;

  char* ws = (char*)d_ws;
  unsigned short* qbf   = (unsigned short*)ws;                                      // 4 MB
  unsigned short* mbf   = (unsigned short*)(ws + (size_t)B_ * D_ * 2);              // 4 MB
  float*          proto = (float*)(ws + (size_t)B_ * D_ * 2 + (size_t)S_ * D_ * 2); // 8 MB

  qprep_kernel<<<B_, 128, 0, stream>>>(x, Wq, bq, qbf);
  mprep_kernel<<<S_, 128, 0, stream>>>(mem, mbf);
  simtopk_kernel<<<B_ / ROWS_PER_WG_, 128, 0, stream>>>(qbf, mbf, mem, proto);

  const size_t total4 = (size_t)B_ * N_ * (D_ / 4);
  addout_kernel<<<(unsigned)((total4 + 255) / 256), 256, 0, stream>>>(x, proto, scale, out);
}
